// AttentionHead_10917806866788
// MI455X (gfx1250) — compile-verified, matrix operand order checked by reading, not run on hardware
//
#include <hip/hip_runtime.h>
#include <math.h>

// ---------------------------------------------------------------------------
// AttentionHead for MI455X (gfx1250, wave32, WMMA 16x16x32 bf16 + TDM)
//   x  [8, 2048, 1024] f32 ; Wq/Wk/Wv [1024, 64] f32  ->  out [8, 2048, 64] f32
// Phase 1: QKV projection (x @ W) in bf16 WMMA, results in d_ws as bf16.
// Phase 2: causal flash attention; K/V tiles staged into LDS by the Tensor
//          Data Mover (tensor_load_to_lds + s_wait_tensorcnt), WMMA for
//          Q@K^T and P@V, online softmax.
// ---------------------------------------------------------------------------

#define BATCH 8
#define SEQ   2048
#define EMBED 1024
#define HDIM  64
#define ROWS  (BATCH * SEQ)          // 16384
#define WT_STRIDE 1032               // padded K-stride for transposed W in LDS

typedef __attribute__((ext_vector_type(16))) __bf16 v16bf;
typedef __attribute__((ext_vector_type(8)))  float  v8f;

// TDM descriptor group types (clang-23 / amdgpu-toolchain 6-arg builtin).
typedef __attribute__((ext_vector_type(4))) unsigned tdm_v4u;
typedef __attribute__((ext_vector_type(8))) int      tdm_v8i;
typedef __attribute__((ext_vector_type(4))) int      tdm_v4i;

// ---- helpers --------------------------------------------------------------

__device__ __forceinline__ __bf16 f32_to_bf16(float f) {
    return (__bf16)f;                 // hardware v_cvt_*_bf16_f32 on gfx1250
}

__device__ __forceinline__ v8f zero_v8f() {
    v8f z = {0.f, 0.f, 0.f, 0.f, 0.f, 0.f, 0.f, 0.f};
    return z;
}

// Load a 16-half fragment as two aligned 16B chunks:
// elements 0..7  <- p[0..7]   (K = base + kb8 + e)
// elements 8..15 <- p[16..23] (K = base + kb8 + 16 + e)
__device__ __forceinline__ v16bf load_frag_pair(const __bf16* p) {
    union { uint4 u[2]; v16bf v; } t;
    t.u[0] = *(const uint4*)(p);
    t.u[1] = *(const uint4*)(p + 16);
    return t.v;
}

// Issue one TDM 2D tile load: global (row-major, bf16) -> LDS.
//   gaddr     : 64-bit global byte address of tile start
//   lds_addr  : byte offset of destination within LDS
//   tile rows = 32 (keys), tile cols = 64 (d), tensor stride = 64 elements
__device__ __forceinline__ void tdm_load_tile_2d(unsigned long long gaddr,
                                                 unsigned lds_addr) {
    // D# group 0: count=1 | lds_addr | global_addr[56:0] | type=2
    tdm_v4u g0 = {
        1u,                                                   // count=1, user
        lds_addr,                                             // bits 63:32
        (unsigned)(gaddr & 0xFFFFFFFFull),                    // ga[31:0]
        (unsigned)((gaddr >> 32) & 0x01FFFFFFull) | 0x80000000u // ga[56:32]|type=2
    };
    // D# group 1 (bit-packed per CDNA5 ISA):
    //   data_size=1 (2 bytes), tensor_dim0=64, tensor_dim1=2048,
    //   tile_dim0=64, tile_dim1=32, tensor_dim0_stride=64.
    const unsigned td0 = HDIM, td1 = SEQ, tile0 = HDIM, tile1 = 32;
    const unsigned stride0 = HDIM;
    tdm_v8i g1 = {
        (int)(1u << 16),                      // [17:16] data_size = 2 bytes
        (int)((td0 & 0xFFFFu) << 16),         // [79:48] tensor_dim0 (lo16)
        (int)((td0 >> 16) | ((td1 & 0xFFFFu) << 16)),   // td0 hi / td1 lo
        (int)((td1 >> 16) | (tile0 << 16)),   // td1 hi / [127:112] tile_dim0
        (int)(tile1),                         // [143:128] tile_dim1
        (int)(stride0),                       // [191:160] dim0_stride lo32
        0,                                    // dim0_stride hi16 / dim1_stride lo16
        0                                     // dim1_stride hi32
    };
    tdm_v4i gz4 = {0, 0, 0, 0};               // 2D tensor: groups 2/3 unused
    tdm_v8i gz8 = {0, 0, 0, 0, 0, 0, 0, 0};   // extra group (6-arg form)
    __builtin_amdgcn_tensor_load_to_lds(g0, g1, gz4, gz4, gz8, 0);
}

// ---------------------------------------------------------------------------
// Kernel 1: QKV projection. grid = (128 row-tiles, 3 weights), block = 256.
// Each wave computes a 16x64 tile of one projection. W lives transposed in LDS
// as bf16 so B-fragments are two contiguous ds_load_b128 per 16x16 tile.
// ---------------------------------------------------------------------------
__global__ void qkv_proj_kernel(const float* __restrict__ x,
                                const float* __restrict__ Wq,
                                const float* __restrict__ Wk,
                                const float* __restrict__ Wv,
                                __bf16* __restrict__ qo,
                                __bf16* __restrict__ ko,
                                __bf16* __restrict__ vo) {
    extern __shared__ __bf16 wT[];   // [HDIM][WT_STRIDE] : W transposed, bf16

    const int which = blockIdx.y;
    const float* __restrict__ W = (which == 0) ? Wq : (which == 1) ? Wk : Wv;
    __bf16* __restrict__ out    = (which == 0) ? qo : (which == 1) ? ko : vo;

    const int tid = threadIdx.x;

    // Stage W (1024x64 f32) -> LDS transposed bf16 [n][k].
    {
        const int n  = tid & 63;         // output column
        const int kc = tid >> 6;         // 0..3, 256 k-values each
        const int k0 = kc * 256;
        #pragma unroll 4
        for (int kk = 0; kk < 256; ++kk) {
            wT[(size_t)n * WT_STRIDE + (k0 + kk)] =
                f32_to_bf16(W[(size_t)(k0 + kk) * HDIM + n]);
        }
    }
    __syncthreads();

    const int wave = tid >> 5;
    const int lane = tid & 31;
    const int m    = lane & 15;          // row within tile (A) / col within tile (B,C)
    const int half = lane >> 4;          // half-wave select
    const int kb8  = half * 8;           // K sub-offset per ISA fragment layout

    const int row0 = blockIdx.x * 128 + wave * 16;
    const float* __restrict__ xrow = x + (size_t)(row0 + m) * EMBED;

    v8f acc[4];
    #pragma unroll
    for (int t = 0; t < 4; ++t) acc[t] = zero_v8f();

    for (int kk = 0; kk < EMBED; kk += 32) {
        // A fragment: 16x32 bf16 slice of x for this wave's 16 rows.
        const float4* p0 = (const float4*)(xrow + kk + kb8);
        const float4* p1 = (const float4*)(xrow + kk + 16 + kb8);
        float4 f0 = p0[0], f1 = p0[1];
        float4 f2 = p1[0], f3 = p1[1];
        v16bf a;
        a[0]  = f32_to_bf16(f0.x); a[1]  = f32_to_bf16(f0.y);
        a[2]  = f32_to_bf16(f0.z); a[3]  = f32_to_bf16(f0.w);
        a[4]  = f32_to_bf16(f1.x); a[5]  = f32_to_bf16(f1.y);
        a[6]  = f32_to_bf16(f1.z); a[7]  = f32_to_bf16(f1.w);
        a[8]  = f32_to_bf16(f2.x); a[9]  = f32_to_bf16(f2.y);
        a[10] = f32_to_bf16(f2.z); a[11] = f32_to_bf16(f2.w);
        a[12] = f32_to_bf16(f3.x); a[13] = f32_to_bf16(f3.y);
        a[14] = f32_to_bf16(f3.z); a[15] = f32_to_bf16(f3.w);

        #pragma unroll
        for (int t = 0; t < 4; ++t) {
            const int n = t * 16 + m;    // output column this lane serves
            v16bf b = load_frag_pair(&wT[(size_t)n * WT_STRIDE + kk + kb8]);
            acc[t] = __builtin_amdgcn_wmma_f32_16x16x32_bf16(
                false, a, false, b, (short)0, acc[t], false, false);
        }
    }

    // Store C/D layout (VGPR r -> rows r / r+8 by half-wave) as bf16.
    #pragma unroll
    for (int t = 0; t < 4; ++t) {
        #pragma unroll
        for (int r = 0; r < 8; ++r) {
            const int row = row0 + half * 8 + r;
            const int col = t * 16 + m;
            out[(size_t)row * HDIM + col] = f32_to_bf16(acc[t][r]);
        }
    }
}

// ---------------------------------------------------------------------------
// Kernel 2: causal flash attention. grid = (S/64, B), block = 128 (4 waves).
// Each wave: 16 query rows x D=64 f32 accumulators, online softmax.
// K/V tiles staged row-major into LDS by the Tensor Data Mover.
// ---------------------------------------------------------------------------
__global__ void flash_attn_kernel(const __bf16* __restrict__ qw,
                                  const __bf16* __restrict__ kw,
                                  const __bf16* __restrict__ vw,
                                  float* __restrict__ out) {
    __shared__ __bf16 Kt[32][HDIM];      // [key][d]  row-major (TDM dest)
    __shared__ __bf16 Vt[32][HDIM];      // [key][d]  row-major (TDM dest)
    __shared__ __bf16 Pt[4][16][32];     // per-wave softmax numerators

    const int tid   = threadIdx.x;
    const int wave  = tid >> 5;
    const int lane  = tid & 31;
    const int m     = lane & 15;
    const int half  = lane >> 4;
    const int kb8   = half * 8;
    const int batch = blockIdx.y;
    const int q0    = blockIdx.x * 64 + wave * 16;   // wave's first query row

    const size_t bbase = (size_t)batch * SEQ * HDIM;
    const __bf16* __restrict__ qrow = qw + bbase + (size_t)(q0 + m) * HDIM;

    // LDS byte offsets of the TDM destinations (low 32 bits of generic ptr).
    const unsigned kt_lds = (unsigned)(size_t)(&Kt[0][0]);
    const unsigned vt_lds = (unsigned)(size_t)(&Vt[0][0]);

    // Q fragments for the whole kernel: 16 rows x 64 d = 2 A-fragments.
    v16bf qa0 = load_frag_pair(qrow + 0  + kb8);
    v16bf qa1 = load_frag_pair(qrow + 32 + kb8);

    v8f acc[4];
    #pragma unroll
    for (int t = 0; t < 4; ++t) acc[t] = zero_v8f();
    float m_run[8], l_run[8];
    #pragma unroll
    for (int r = 0; r < 8; ++r) { m_run[r] = -1e30f; l_run[r] = 0.f; }

    const float scale = 0.125f;                        // 1/sqrt(64)
    const int n_kblocks = (blockIdx.x + 1) * 2;        // causal: keys <= tile end

    for (int kb = 0; kb < n_kblocks; ++kb) {
        const int kbase = kb * 32;

        // ---- TDM stage of K and V 32x64 bf16 tiles (one issuing wave) ----
        if (tid == 0) {
            const unsigned long long kga =
                (unsigned long long)(size_t)(kw + bbase + (size_t)kbase * HDIM);
            const unsigned long long vga =
                (unsigned long long)(size_t)(vw + bbase + (size_t)kbase * HDIM);
            tdm_load_tile_2d(kga, kt_lds);
            tdm_load_tile_2d(vga, vt_lds);
            __builtin_amdgcn_s_wait_tensorcnt(0);
        }
        __syncthreads();

        // ---- scores: two 16x16 tiles (keys j*16..j*16+15), 2 WMMAs each ----
        float p[2][8];
        float tmax[8];
        #pragma unroll
        for (int r = 0; r < 8; ++r) tmax[r] = -1e30f;

        float sv[2][8];
        #pragma unroll
        for (int j = 0; j < 2; ++j) {
            const __bf16* kp = &Kt[j * 16 + m][0];
            v16bf b0 = load_frag_pair(kp + 0  + kb8);
            v16bf b1 = load_frag_pair(kp + 32 + kb8);
            v8f sc = zero_v8f();
            sc = __builtin_amdgcn_wmma_f32_16x16x32_bf16(
                false, qa0, false, b0, (short)0, sc, false, false);
            sc = __builtin_amdgcn_wmma_f32_16x16x32_bf16(
                false, qa1, false, b1, (short)0, sc, false, false);

            const int key = kbase + j * 16 + m;        // this lane's key column
            #pragma unroll
            for (int r = 0; r < 8; ++r) {
                const int row = q0 + half * 8 + r;
                float s = sc[r] * scale;
                if (key > row) s = -1e30f;             // causal mask
                sv[j][r] = s;
                tmax[r] = fmaxf(tmax[r], s);
            }
        }

        // row-max across the 16 lanes sharing each row (stays in half-wave)
        #pragma unroll
        for (int off = 1; off < 16; off <<= 1) {
            #pragma unroll
            for (int r = 0; r < 8; ++r)
                tmax[r] = fmaxf(tmax[r], __shfl_xor(tmax[r], off, 32));
        }

        // online softmax update
        float rsum[8];
        #pragma unroll
        for (int r = 0; r < 8; ++r) {
            const float mnew = fmaxf(m_run[r], tmax[r]);
            const float corr = __expf(m_run[r] - mnew);
            p[0][r] = __expf(sv[0][r] - mnew);
            p[1][r] = __expf(sv[1][r] - mnew);
            rsum[r] = p[0][r] + p[1][r];
            m_run[r] = mnew;
            l_run[r] *= corr;
            #pragma unroll
            for (int t = 0; t < 4; ++t) acc[t][r] *= corr;
        }
        #pragma unroll
        for (int off = 1; off < 16; off <<= 1) {
            #pragma unroll
            for (int r = 0; r < 8; ++r)
                rsum[r] += __shfl_xor(rsum[r], off, 32);
        }
        #pragma unroll
        for (int r = 0; r < 8; ++r) l_run[r] += rsum[r];

        // stage P (C-layout -> LDS) so it can be re-read in A-fragment layout
        #pragma unroll
        for (int j = 0; j < 2; ++j)
            #pragma unroll
            for (int r = 0; r < 8; ++r)
                Pt[wave][half * 8 + r][j * 16 + m] = f32_to_bf16(p[j][r]);
        __syncthreads();

        // ---- PV: out(16x64) += P(16x32) @ V(32x64), 4 WMMAs ----
        // V B-fragment: lane serves column d = t*16+m; gather keys kb8+e /
        // 16+kb8+e from the row-major V tile (strided ds_load_u16).
        v16bf pa = load_frag_pair(&Pt[wave][m][kb8]);
        #pragma unroll
        for (int t = 0; t < 4; ++t) {
            const int d = t * 16 + m;
            v16bf vbfrag;
            #pragma unroll
            for (int e = 0; e < 8; ++e) vbfrag[e]     = Vt[kb8 + e][d];
            #pragma unroll
            for (int e = 0; e < 8; ++e) vbfrag[8 + e] = Vt[16 + kb8 + e][d];
            acc[t] = __builtin_amdgcn_wmma_f32_16x16x32_bf16(
                false, pa, false, vbfrag, (short)0, acc[t], false, false);
        }
        __syncthreads();   // protect Kt/Vt before next block's TDM refill
    }

    // ---- normalize and store f32 output ----
    #pragma unroll
    for (int t = 0; t < 4; ++t) {
        #pragma unroll
        for (int r = 0; r < 8; ++r) {
            const int row = q0 + half * 8 + r;
            const int col = t * 16 + m;
            const float inv = 1.0f / l_run[r];
            out[bbase + (size_t)row * HDIM + col] = acc[t][r] * inv;
        }
    }
}

// ---------------------------------------------------------------------------

extern "C" void kernel_launch(void* const* d_in, const int* in_sizes, int n_in,
                              void* d_out, int out_size, void* d_ws, size_t ws_size,
                              hipStream_t stream) {
    (void)in_sizes; (void)n_in; (void)out_size; (void)ws_size;

    const float* x  = (const float*)d_in[0];
    const float* Wq = (const float*)d_in[1];
    const float* Wk = (const float*)d_in[2];
    const float* Wv = (const float*)d_in[3];

    __bf16* qw = (__bf16*)d_ws;
    __bf16* kw = qw + (size_t)ROWS * HDIM;
    __bf16* vw = kw + (size_t)ROWS * HDIM;

    // Phase 1: QKV projections (128 row-tiles x {Q,K,V}); W transposed in LDS.
    const size_t lds1 = (size_t)HDIM * WT_STRIDE * sizeof(__bf16);
    dim3 g1(ROWS / 128, 3);
    qkv_proj_kernel<<<g1, 256, lds1, stream>>>(x, Wq, Wk, Wv, qw, kw, vw);

    // Phase 2: causal flash attention, 64-query tiles per block.
    dim3 g2(SEQ / 64, BATCH);
    flash_attn_kernel<<<g2, 128, 0, stream>>>(qw, kw, vw, (float*)d_out);
}
